// NTXentLoss_54640573939753
// MI455X (gfx1250) — matrix operand order repaired and checked by reading, not compiled, not run on hardware
//
#include <hip/hip_runtime.h>
#include <hip/hip_fp16.h>

typedef __attribute__((ext_vector_type(16))) _Float16 v16h;
typedef __attribute__((ext_vector_type(8)))  _Float16 v8h;
typedef __attribute__((ext_vector_type(4)))  _Float16 v4h;
typedef __attribute__((ext_vector_type(8)))  float    v8f;

#define N2 16384          // number of rows (2N)
#define DIM 128           // feature dim
#define NSPLIT 8          // column splits
#define COLS_PER_SPLIT (N2 / NSPLIT)          // 2048
#define TILES_PER_SPLIT (COLS_PER_SPLIT / 16) // 128
#define STRIPES 4                             // 64 rows per wave
#define ROWS_PER_BLOCK (8 * STRIPES * 16)     // 512

union ABFrag { v16h v; v8h h[2]; };

// ---------------------------------------------------------------------------
// Kernel 1: convert z (f32) -> z16 (f16). 4 elements per thread, vectorized.
// ---------------------------------------------------------------------------
__global__ __launch_bounds__(256)
void ntxent_prep(const float* __restrict__ z, _Float16* __restrict__ z16) {
  int i = (blockIdx.x * 256 + threadIdx.x) * 4;
  float4 f = *(const float4*)(z + i);
  v4h o = {(_Float16)f.x, (_Float16)f.y, (_Float16)f.z, (_Float16)f.w};
  *(v4h*)(z16 + i) = o;
}

// ---------------------------------------------------------------------------
// Kernel 2: exact positives in f32: P[i] = 2 * dot(z[i], z[(i+N)%2N])
// One wave per row; 8 waves per block.
// ---------------------------------------------------------------------------
__global__ __launch_bounds__(256)
void ntxent_pos(const float* __restrict__ z, float* __restrict__ P) {
  int lane = threadIdx.x & 31;
  int wave = threadIdx.x >> 5;
  int row  = blockIdx.x * 8 + wave;
  int prow = (row + N2 / 2) & (N2 - 1);
  const float* a = z + (size_t)row  * DIM;
  const float* b = z + (size_t)prow * DIM;
  float s = 0.f;
#pragma unroll
  for (int j = 0; j < DIM / 32; j++)
    s += a[lane + j * 32] * b[lane + j * 32];
#pragma unroll
  for (int m = 1; m < 32; m <<= 1) s += __shfl_xor(s, m, 32);
  if (lane == 0) P[row] = 2.0f * s;   // sim/T for the positive pair
}

// ---------------------------------------------------------------------------
// Kernel 3: fused  z16 @ z16^T  +  streaming sum of exp(sim/T - 2).
// Each wave: 64 rows (4 WMMA stripes, A held in 128 VGPRs) x 2048 columns.
// 16 WMMAs per 8 B-loads; 4 independent accumulator chains for ILP.
// Block: 8 waves = 512 rows.  Grid: (32 row-blocks, 8 column-splits).
// Partial sums written to Spart[row*NSPLIT + split] (no atomics).
// ---------------------------------------------------------------------------
__global__ __launch_bounds__(256)
void ntxent_main(const _Float16* __restrict__ z16, float* __restrict__ Spart) {
  const int lane  = threadIdx.x & 31;
  const int wave  = threadIdx.x >> 5;
  const int l15   = lane & 15;
  const int half8 = (lane >> 4) << 3;          // 0 for lanes 0-15, 8 for 16-31

  const int rowBase = blockIdx.x * ROWS_PER_BLOCK + wave * (STRIPES * 16);
  const int colBase = blockIdx.y * COLS_PER_SPLIT;

  // --- load A fragments for 4 stripes x 4 k-steps (held for whole loop) ---
  // A 16x32 f16 layout: lane holds row (l15); halves 0-7 = K + (0|8),
  // halves 8-15 = K + 16 + (0|8). All contiguous 16B loads.
  v16h a[STRIPES][4];
#pragma unroll
  for (int s = 0; s < STRIPES; s++) {
    const _Float16* ap = z16 + (size_t)(rowBase + s * 16 + l15) * DIM;
#pragma unroll
    for (int kk = 0; kk < 4; kk++) {
      ABFrag u;
      u.h[0] = *(const v8h*)(ap + kk * 32 + half8);
      u.h[1] = *(const v8h*)(ap + kk * 32 + 16 + half8);
      a[s][kk] = u.v;
    }
  }

  v8f sacc[STRIPES];
#pragma unroll
  for (int s = 0; s < STRIPES; s++)
    sacc[s] = (v8f){0.f, 0.f, 0.f, 0.f, 0.f, 0.f, 0.f, 0.f};

  const float C1 = 2.8853900817779268f;   // 2*log2(e): exp(2x-2)=exp2((x-1)*C1)

  for (int t = 0; t < TILES_PER_SPLIT; t++) {
    const int c = colBase + t * 16;

    // --- B fragments: B = z^T, so column (c+l15) of B is row (c+l15) of z.
    // B 32x16 f16 layout: halves 0-7 = K + (0|8), halves 8-15 = K + 16 + (0|8).
    const _Float16* bp = z16 + (size_t)(c + l15) * DIM + (lane < 16 ? 0 : 8);
    v16h b[4];
#pragma unroll
    for (int kk = 0; kk < 4; kk++) {
      ABFrag u;
      u.h[0] = *(const v8h*)(bp + kk * 32);
      u.h[1] = *(const v8h*)(bp + kk * 32 + 16);
      b[kk] = u.v;
    }

    if (t + 1 < TILES_PER_SPLIT)
      __builtin_prefetch(z16 + (size_t)(c + 16 + l15) * DIM, 0, 3);

    // --- 16x16x128 dot via 4 chained WMMAs per stripe, 4 parallel chains ---
    v8f acc[STRIPES];
#pragma unroll
    for (int s = 0; s < STRIPES; s++)
      acc[s] = (v8f){0.f, 0.f, 0.f, 0.f, 0.f, 0.f, 0.f, 0.f};
#pragma unroll
    for (int kk = 0; kk < 4; kk++) {
#pragma unroll
      for (int s = 0; s < STRIPES; s++) {
        acc[s] = __builtin_amdgcn_wmma_f32_16x16x32_f16(
            false, a[s][kk], false, b[kk], (short)0, acc[s], false, false);
      }
    }

    // --- streaming exp-sum with fixed shift 2 (sim/T is bounded by 2).
    // C/D layout: lane has column N=l15, rows v+half8 (v=0..7).
    // Diagonal (row==col) only possible in the tile where c == stripe base.
#pragma unroll
    for (int s = 0; s < STRIPES; s++) {
      const bool diag = (c == rowBase + s * 16);
#pragma unroll
      for (int v = 0; v < 8; v++) {
        float e = __builtin_amdgcn_exp2f(__builtin_fmaf(acc[s][v], C1, -C1));
        if (diag && (v + half8 == l15)) e = 0.f;  // masked self-similarity
        sacc[s][v] += e;
      }
    }
  }

  // --- reduce across the 16 lanes holding each row's columns ---
#pragma unroll
  for (int m = 1; m < 16; m <<= 1) {
#pragma unroll
    for (int s = 0; s < STRIPES; s++) {
#pragma unroll
      for (int v = 0; v < 8; v++)
        sacc[s][v] += __shfl_xor(sacc[s][v], m, 32);
    }
  }
  if (l15 == 0) {   // lane 0 -> rows base+0..7, lane 16 -> rows base+8..15
#pragma unroll
    for (int s = 0; s < STRIPES; s++) {
      const int r = rowBase + s * 16 + half8;
#pragma unroll
      for (int v = 0; v < 8; v++)
        Spart[(size_t)(r + v) * NSPLIT + blockIdx.y] = sacc[s][v];
    }
  }
}

// ---------------------------------------------------------------------------
// Kernel 4: loss = mean_i( -P[i] + log(sum_k Spart[i][k]) + 2 )
// ---------------------------------------------------------------------------
__global__ __launch_bounds__(256)
void ntxent_final(const float* __restrict__ Spart, const float* __restrict__ P,
                  float* __restrict__ out) {
  __shared__ float red[256];
  float local = 0.f;
  for (int i = threadIdx.x; i < N2; i += 256) {
    float S = 0.f;
#pragma unroll
    for (int k = 0; k < NSPLIT; k++) S += Spart[(size_t)i * NSPLIT + k];
    // ln(S) = log2(S) * ln(2); shift of 2 restored here
    local += 2.0f + 0.69314718055994531f * __builtin_amdgcn_logf(S) - P[i];
  }
  red[threadIdx.x] = local;
  __syncthreads();
  for (int st = 128; st > 0; st >>= 1) {
    if (threadIdx.x < st) red[threadIdx.x] += red[threadIdx.x + st];
    __syncthreads();
  }
  if (threadIdx.x == 0) out[0] = red[0] / (float)N2;
}

// ---------------------------------------------------------------------------
extern "C" void kernel_launch(void* const* d_in, const int* in_sizes, int n_in,
                              void* d_out, int out_size, void* d_ws, size_t ws_size,
                              hipStream_t stream) {
  (void)in_sizes; (void)n_in; (void)out_size; (void)ws_size;
  const float* z = (const float*)d_in[0];
  float* out = (float*)d_out;

  // workspace layout: z16 (4MB) | Spart (512KB) | P (64KB)
  char* ws = (char*)d_ws;
  _Float16* z16 = (_Float16*)ws;
  float* Spart  = (float*)(ws + (size_t)N2 * DIM * 2);
  float* P      = (float*)(ws + (size_t)N2 * DIM * 2 + (size_t)N2 * NSPLIT * 4);

  ntxent_prep<<<(N2 * DIM / 4) / 256, 256, 0, stream>>>(z, z16);
  ntxent_pos<<<N2 / 8, 256, 0, stream>>>(z, P);
  ntxent_main<<<dim3(N2 / ROWS_PER_BLOCK, NSPLIT), 256, 0, stream>>>(z16, Spart);
  ntxent_final<<<1, 256, 0, stream>>>(Spart, P, out);
}
